// Encoder_91096256348319
// MI455X (gfx1250) — matrix operand order repaired and checked by reading, not compiled, not run on hardware
//
#include <hip/hip_runtime.h>

// ---------------------------------------------------------------------------
// Bidirectional masked GRU encoder for MI455X (gfx1250, wave32, WMMA).
//   B=32, T=512, E=256, U=512, V=32000
// Plan:
//   K0: transpose+convert weights fp32 -> bf16 (W: [E,3U]->[3U,E], U: [U,3U]->[3U,U])
//   K1: embedding gather + x@W + b_in via v_wmma_f32_16x16x32_bf16 (both dirs)
//   K2: persistent recurrence: 32 WGs (16/dir), each owns 32 hidden units;
//       its 96x512 bf16 slice of U lives in LDS for all 512 steps; h state is
//       exchanged via ping-pong bf16 buffers in L2 + one grid barrier / step.
// CDNA5 paths: v_wmma_f32_16x16x32_bf16, global_load_async_to_lds_b128
//   (ASYNCcnt) for LDS staging, global_prefetch for next-step xproj rows.
// ---------------------------------------------------------------------------

typedef __bf16 v8bf  __attribute__((ext_vector_type(8)));
typedef __bf16 v16bf __attribute__((ext_vector_type(16)));
typedef float  v8f   __attribute__((ext_vector_type(8)));
typedef int    v4i   __attribute__((ext_vector_type(4)));

// ---- async global->LDS staging (CDNA5), with portable fallbacks -----------
#if __has_builtin(__builtin_amdgcn_global_load_async_to_lds_b128)
#define HAVE_ASYNC_LDS 1
#else
#define HAVE_ASYNC_LDS 0
#endif

// builtin signature (from clang diagnostics): (v4i AS1*, v4i AS3*, imm, imm)
#define AS_G_V4I(p) ((__attribute__((address_space(1))) v4i*)(uintptr_t)(p))
#define AS_L_V4I(p) ((__attribute__((address_space(3))) v4i*)(uintptr_t)(p))

__device__ __forceinline__ void wait_async_zero() {
#if __has_builtin(__builtin_amdgcn_s_wait_asynccnt)
  __builtin_amdgcn_s_wait_asynccnt(0);
#else
  asm volatile("s_wait_asynccnt 0" ::: "memory");
#endif
}

// copy `n16` 16-byte chunks global->LDS, strided across the block
__device__ __forceinline__ void stage_to_lds_b128(void* lds_dst,
                                                  const void* gsrc, int n16,
                                                  int tid, int nthr) {
#if HAVE_ASYNC_LDS
  for (int c = tid; c < n16; c += nthr)
    __builtin_amdgcn_global_load_async_to_lds_b128(AS_G_V4I(gsrc) + c,
                                                   AS_L_V4I(lds_dst) + c, 0,
                                                   0);
  wait_async_zero();
#else
  for (int c = tid; c < n16; c += nthr)
    ((uint4*)lds_dst)[c] = ((const uint4*)gsrc)[c];
#endif
}

namespace {
constexpr int kB  = 32;
constexpr int kT  = 512;
constexpr int kE  = 256;
constexpr int kU  = 512;
constexpr int kN3 = 3 * kU;          // 1536
constexpr int kM  = kB * kT;         // 16384

// workspace layout (bytes)
constexpr size_t OFF_BAR  = 0;                         // 16 uints (barriers)
constexpr size_t OFF_H    = 64;                        // 4 x bf16[32*512] (dir x pingpong)
constexpr size_t HELEMS   = (size_t)kB * kU;           // 16384
constexpr size_t ZERO_BYTES = OFF_H + 4 * HELEMS * 2;  // 131136
constexpr size_t OFF_WT_F = 131328;
constexpr size_t WT_SZ    = (size_t)kN3 * kE * 2;      // 786432
constexpr size_t OFF_WT_B = OFF_WT_F + WT_SZ;
constexpr size_t OFF_UT_F = OFF_WT_B + WT_SZ;
constexpr size_t UT_SZ    = (size_t)kN3 * kU * 2;      // 1572864
constexpr size_t OFF_UT_B = OFF_UT_F + UT_SZ;
constexpr size_t OFF_XP_F = OFF_UT_B + UT_SZ;
constexpr size_t XP_SZ    = (size_t)kM * kN3 * 4;      // 100663296
constexpr size_t OFF_XP_B = OFF_XP_F + XP_SZ;
} // namespace

__device__ __forceinline__ v16bf cat8(v8bf a, v8bf b) {
  return __builtin_shufflevector(a, b, 0, 1, 2, 3, 4, 5, 6, 7,
                                 8, 9, 10, 11, 12, 13, 14, 15);
}

__device__ __forceinline__ v8f wmma_bf16(v16bf a, v16bf b, v8f c) {
  // D = A(16x32 bf16) * B(32x16 bf16) + C(16x16 f32)
  return __builtin_amdgcn_wmma_f32_16x16x32_bf16(false, a, false, b,
                                                 (short)0, c, false, false);
}

__device__ __forceinline__ v8f zero8() {
  v8f z = {0.f, 0.f, 0.f, 0.f, 0.f, 0.f, 0.f, 0.f};
  return z;
}

// ---------------------------------------------------------------------------
// K0: out[n*K + k] = bf16(in[k*N + n])  (transpose + convert)
// ---------------------------------------------------------------------------
__global__ void convert_transpose_bf16(const float* __restrict__ in,
                                       __bf16* __restrict__ out,
                                       int K, int N) {
  size_t idx = (size_t)blockIdx.x * blockDim.x + threadIdx.x;
  size_t total = (size_t)K * N;
  if (idx >= total) return;
  int n = (int)(idx / K);
  int k = (int)(idx % K);
  out[idx] = (__bf16)in[(size_t)k * N + n];
}

// ---------------------------------------------------------------------------
// K1: xproj[d] = (emb[tokens] @ W_d) + b_in_d     (WMMA bf16)
// grid: (M/128, N3/128, 2), block 256 (8 waves; 4 along M x 2 along N)
// ---------------------------------------------------------------------------
__global__ __launch_bounds__(256, 2) void xproj_wmma(
    const int* __restrict__ tokens, const float* __restrict__ emb,
    const __bf16* __restrict__ WtF, const __bf16* __restrict__ WtB,
    const float* __restrict__ binF, const float* __restrict__ binB,
    float* __restrict__ xpF, float* __restrict__ xpB) {
  __shared__ __align__(16) __bf16 Atile[128][32];  // [M][K] bf16
  __shared__ __align__(16) __bf16 Btile[128][32];  // [N][K] bf16 (W transposed)
  __shared__ int toks[128];

  const int tid = threadIdx.x;
  const int z = blockIdx.z;
  const __bf16* Wt = z ? WtB : WtF;   // [1536][256]
  const float* bin = z ? binB : binF;
  float* xp = z ? xpB : xpF;

  const int mBlock = blockIdx.x * 128;
  const int nBlock = blockIdx.y * 128;

  if (tid < 128) toks[tid] = tokens[mBlock + tid];
  __syncthreads();

  const int lane = tid & 31;
  const int wid = tid >> 5;
  const int mL = (wid & 3) * 32;  // wave M base in tile
  const int nL = (wid >> 2) * 64; // wave N base in tile
  const int lrow = lane & 15;
  const int aK = (lane < 16) ? 0 : 8;   // ISA A-operand K grouping
  const int bK = (lane < 16) ? 0 : 8;   // ISA B-operand K grouping

  v8f acc[2][4];
#pragma unroll
  for (int mi = 0; mi < 2; ++mi)
#pragma unroll
    for (int ni = 0; ni < 4; ++ni) acc[mi][ni] = zero8();

  const int row = tid >> 1;
  const int half = (tid & 1) * 16;

  for (int kb = 0; kb < kE; kb += 32) {
    // B: bf16 weight panel, async global->LDS (rows = N, contiguous K).
    // Each 128-row x 32-col chunk is row-fragmented in global (row stride
    // kE elems), so issue one b128 per 8-elem group per thread.
#if HAVE_ASYNC_LDS
    {
      const __bf16* src = Wt + (size_t)(nBlock + row) * kE + kb + half;
      __bf16* dst = &Btile[row][half];
      __builtin_amdgcn_global_load_async_to_lds_b128(AS_G_V4I(src),
                                                     AS_L_V4I(dst), 0, 0);
      __builtin_amdgcn_global_load_async_to_lds_b128(AS_G_V4I(src + 8),
                                                     AS_L_V4I(dst + 8), 0, 0);
    }
#else
    {
      const uint4* src =
          (const uint4*)(Wt + (size_t)(nBlock + row) * kE + kb + half);
      uint4 w0 = src[0], w1 = src[1];
      *(uint4*)&Btile[row][half] = w0;
      *(uint4*)&Btile[row][half + 8] = w1;
    }
#endif
    // A: gather embedding rows, convert fp32 -> bf16 (needs VALU, stays manual)
    {
      const float4* src =
          (const float4*)(emb + (size_t)toks[row] * kE + kb + half);
      float4 f0 = src[0], f1 = src[1], f2 = src[2], f3 = src[3];
      float fv[16] = {f0.x, f0.y, f0.z, f0.w, f1.x, f1.y, f1.z, f1.w,
                      f2.x, f2.y, f2.z, f2.w, f3.x, f3.y, f3.z, f3.w};
      v8bf p0, p1;
#pragma unroll
      for (int j = 0; j < 8; ++j) {
        p0[j] = (__bf16)fv[j];
        p1[j] = (__bf16)fv[j + 8];
      }
      *(v8bf*)&Atile[row][half] = p0;
      *(v8bf*)&Atile[row][half + 8] = p1;
    }
#if HAVE_ASYNC_LDS
    wait_async_zero();
#endif
    __syncthreads();

    v16bf af[2], bf_[4];
#pragma unroll
    for (int mi = 0; mi < 2; ++mi) {
      const __bf16* ap = &Atile[mL + mi * 16 + lrow][aK];
      af[mi] = cat8(*(const v8bf*)ap, *(const v8bf*)(ap + 16));
    }
#pragma unroll
    for (int ni = 0; ni < 4; ++ni) {
      const __bf16* bp = &Btile[nL + ni * 16 + lrow][bK];
      bf_[ni] = cat8(*(const v8bf*)bp, *(const v8bf*)(bp + 16));
    }
#pragma unroll
    for (int mi = 0; mi < 2; ++mi)
#pragma unroll
      for (int ni = 0; ni < 4; ++ni)
        acc[mi][ni] = wmma_bf16(af[mi], bf_[ni], acc[mi][ni]);

    __syncthreads();
  }

  // epilogue: + b_in, store fp32
  const int lhi = (lane >> 4) * 8;
#pragma unroll
  for (int mi = 0; mi < 2; ++mi) {
#pragma unroll
    for (int ni = 0; ni < 4; ++ni) {
      const int gn = nBlock + nL + ni * 16 + lrow;
      const float bv = bin[gn];
#pragma unroll
      for (int r = 0; r < 8; ++r) {
        const int gm = mBlock + mL + mi * 16 + r + lhi;
        xp[(size_t)gm * kN3 + gn] = acc[mi][ni][r] + bv;
      }
    }
  }
}

// ---------------------------------------------------------------------------
// device-scope sense barrier over the 16 blocks of one direction
// ---------------------------------------------------------------------------
__device__ __forceinline__ void grid_sync(unsigned* counter, unsigned* gen,
                                          unsigned nblocks) {
  __threadfence();  // make this block's global h writes visible
  __syncthreads();
  if (threadIdx.x == 0) {
    unsigned g =
        __hip_atomic_load(gen, __ATOMIC_RELAXED, __HIP_MEMORY_SCOPE_AGENT);
    unsigned arrived = __hip_atomic_fetch_add(counter, 1u, __ATOMIC_ACQ_REL,
                                              __HIP_MEMORY_SCOPE_AGENT);
    if (arrived + 1 == nblocks) {
      __hip_atomic_store(counter, 0u, __ATOMIC_RELAXED,
                         __HIP_MEMORY_SCOPE_AGENT);
      __hip_atomic_fetch_add(gen, 1u, __ATOMIC_RELEASE,
                             __HIP_MEMORY_SCOPE_AGENT);
    } else {
      while (__hip_atomic_load(gen, __ATOMIC_ACQUIRE,
                               __HIP_MEMORY_SCOPE_AGENT) == g) {
        __builtin_amdgcn_s_sleep(2);
      }
    }
  }
  __syncthreads();
}

// ---------------------------------------------------------------------------
// K2: persistent recurrence. grid = 32 blocks x 128 threads (4 waves).
//   block = (dir, g): dir = bid>>4, g = bid&15; owns hidden cols [g*32, g*32+32)
//   LDS: U panel [96][512] bf16 (resident all steps) + h [32][512] bf16
//        + rec [32][96] f32 + h_own [32][32] f32   = 147456 B dynamic LDS
//   per step: rec = h @ U_panel  (2 Mtiles x 6 Ntiles x 16 Ktiles WMMA),
//             gates, masked update, write seq out + bf16 h to ping-pong buf.
// ---------------------------------------------------------------------------
__global__ __launch_bounds__(128, 1) void gru_persistent(
    const int* __restrict__ tokens, const float* __restrict__ xpF,
    const float* __restrict__ xpB, const __bf16* __restrict__ UtF,
    const __bf16* __restrict__ UtB, const float* __restrict__ brF,
    const float* __restrict__ brB, __bf16* __restrict__ hbuf,
    float* __restrict__ out, unsigned* __restrict__ bar) {
  extern __shared__ __align__(16) char smem[];
  __bf16* Upanel = (__bf16*)smem;                        // [96][512]
  __bf16* h_lds = (__bf16*)(smem + 98304);               // [32][512]
  float* rec = (float*)(smem + 98304 + 32768);           // [32][96]
  float* h_own = (float*)(smem + 98304 + 32768 + 12288); // [32][32]

  const int tid = threadIdx.x;
  const int bid = blockIdx.x;
  const int dir = bid >> 4;
  const int g = bid & 15;

  const float* xp = dir ? xpB : xpF;
  const float* brec = dir ? brB : brF;
  const __bf16* Ut = dir ? UtB : UtF;  // [1536][512] (N-major, contiguous K)

  // --- load this block's 96x512 U panel into LDS (resident for all steps) ---
  // rows of the panel are contiguous 1KB runs in global; async-copy per row.
#if HAVE_ASYNC_LDS
  for (int c = tid; c < 96 * 64; c += 128) {  // 64 b128 per 512-elem row
    const int rloc = c >> 6;
    const int cc = c & 63;
    const int gate = rloc >> 5;
    const int gcol = gate * kU + g * 32 + (rloc & 31);
    __builtin_amdgcn_global_load_async_to_lds_b128(
        AS_G_V4I(Ut + (size_t)gcol * kU) + cc,
        AS_L_V4I(Upanel + (size_t)rloc * kU) + cc, 0, 0);
  }
  wait_async_zero();
#else
  for (int c = tid; c < 96 * 64; c += 128) {
    const int rloc = c >> 6;
    const int cc = c & 63;
    const int gate = rloc >> 5;
    const int gcol = gate * kU + g * 32 + (rloc & 31);
    ((uint4*)(Upanel + (size_t)rloc * kU))[cc] =
        ((const uint4*)(Ut + (size_t)gcol * kU))[cc];
  }
#endif
  for (int i = tid; i < kB * 32; i += 128) h_own[i] = 0.f;
  __syncthreads();

  const int lane = tid & 31;
  const int wid = tid >> 5;
  const int mtile = wid & 1;        // batch rows mtile*16..+16
  const int ntB = (wid >> 1) * 3;   // ntiles ntB..ntB+2 (of 6)
  const int lrow = lane & 15;
  const int aK = (lane < 16) ? 0 : 8;
  const int bK = (lane < 16) ? 0 : 8;
  const int lhi = (lane >> 4) * 8;

  int pp = 0;
  for (int it = 0; it < kT; ++it) {
    const int t = dir ? (kT - 1 - it) : it;

    // --- stage full h_{t-1} (bf16, 32KB) from L2 ping buffer (ASYNCcnt) ---
    const __bf16* hcur = hbuf + (size_t)(dir * 2 + pp) * HELEMS;
    stage_to_lds_b128(h_lds, hcur, 2048, tid, 128);
    __syncthreads();

    // --- rec = h @ Upanel : 3 WMMA tiles per wave, K = 512 ---
    v8f acc[3];
    acc[0] = zero8();
    acc[1] = zero8();
    acc[2] = zero8();
#pragma unroll 4
    for (int kt = 0; kt < 16; ++kt) {
      const int kb = kt * 32;
      const __bf16* ap = h_lds + (size_t)(mtile * 16 + lrow) * kU + kb + aK;
      const v16bf af = cat8(*(const v8bf*)ap, *(const v8bf*)(ap + 16));
#pragma unroll
      for (int j = 0; j < 3; ++j) {
        const __bf16* bp =
            Upanel + (size_t)((ntB + j) * 16 + lrow) * kU + kb + bK;
        const v16bf bf_ = cat8(*(const v8bf*)bp, *(const v8bf*)(bp + 16));
        acc[j] = wmma_bf16(af, bf_, acc[j]);
      }
    }
#pragma unroll
    for (int j = 0; j < 3; ++j) {
      const int n = (ntB + j) * 16 + lrow;
#pragma unroll
      for (int r = 0; r < 8; ++r)
        rec[(mtile * 16 + r + lhi) * 96 + n] = acc[j][r];
    }
    __syncthreads();

    // --- gates + masked state update for the 32x32 owned (b,u) slice ---
    __bf16* hnext = hbuf + (size_t)(dir * 2 + (pp ^ 1)) * HELEMS;
    const int tnext = dir ? (t - 1) : (t + 1);
    const bool has_next = (tnext >= 0) && (tnext < kT);
    for (int e = tid; e < kB * 32; e += 128) {
      const int b = e >> 5;
      const int u = e & 31;
      const int ucol = g * 32 + u;
      const size_t xbase = ((size_t)(b * kT + t)) * kN3;
      const float xz = xp[xbase + ucol];
      const float xr = xp[xbase + kU + ucol];
      const float xh = xp[xbase + 2 * kU + ucol];
      // prefetch next step's xproj rows into cache while WMMA/barrier runs
      if (has_next) {
        const size_t xnext = ((size_t)(b * kT + tnext)) * kN3;
        __builtin_prefetch(&xp[xnext + ucol], 0, 1);
        __builtin_prefetch(&xp[xnext + kU + ucol], 0, 1);
        __builtin_prefetch(&xp[xnext + 2 * kU + ucol], 0, 1);
      }
      const float hz = rec[b * 96 + u] + brec[ucol];
      const float hr = rec[b * 96 + 32 + u] + brec[kU + ucol];
      const float hh = rec[b * 96 + 64 + u] + brec[2 * kU + ucol];
      const float hprev = h_own[b * 32 + u];
      const float zg = 1.f / (1.f + __expf(-(xz + hz)));
      const float rg = 1.f / (1.f + __expf(-(xr + hr)));
      const float cand = tanhf(xh + rg * hh);
      const float hnew = zg * hprev + (1.f - zg) * cand;
      const bool mt = tokens[b * kT + t] != 0;
      const float hout = mt ? hnew : hprev;
      h_own[b * 32 + u] = hout;
      out[((size_t)(b * kT + t)) * (2 * kU) + dir * kU + ucol] = hout;
      hnext[b * kU + ucol] = (__bf16)hout;
    }

    grid_sync(&bar[dir * 2], &bar[dir * 2 + 1], 16);
    pp ^= 1;
  }
}

// ---------------------------------------------------------------------------
extern "C" void kernel_launch(void* const* d_in, const int* in_sizes, int n_in,
                              void* d_out, int out_size, void* d_ws,
                              size_t ws_size, hipStream_t stream) {
  (void)in_sizes; (void)n_in; (void)out_size; (void)ws_size;
  const int* tokens = (const int*)d_in[0];
  const float* emb = (const float*)d_in[1];
  const float* W_fw = (const float*)d_in[2];
  const float* U_fw = (const float*)d_in[3];
  const float* b_in_fw = (const float*)d_in[4];
  const float* b_rec_fw = (const float*)d_in[5];
  const float* W_bw = (const float*)d_in[6];
  const float* U_bw = (const float*)d_in[7];
  const float* b_in_bw = (const float*)d_in[8];
  const float* b_rec_bw = (const float*)d_in[9];
  float* out = (float*)d_out;
  char* ws = (char*)d_ws;

  unsigned* bar = (unsigned*)(ws + OFF_BAR);
  __bf16* hbuf = (__bf16*)(ws + OFF_H);
  __bf16* WtF = (__bf16*)(ws + OFF_WT_F);
  __bf16* WtB = (__bf16*)(ws + OFF_WT_B);
  __bf16* UtF = (__bf16*)(ws + OFF_UT_F);
  __bf16* UtB = (__bf16*)(ws + OFF_UT_B);
  float* xpF = (float*)(ws + OFF_XP_F);
  float* xpB = (float*)(ws + OFF_XP_B);

  // zero barriers + h ping-pong state every call (capture-safe, deterministic)
  (void)hipMemsetAsync(ws, 0, ZERO_BYTES, stream);

  // K0: weight transpose+convert to bf16 (N-major, contiguous K)
  {
    const int nW = kE * kN3, nU = kU * kN3;
    convert_transpose_bf16<<<(nW + 255) / 256, 256, 0, stream>>>(W_fw, WtF, kE, kN3);
    convert_transpose_bf16<<<(nW + 255) / 256, 256, 0, stream>>>(W_bw, WtB, kE, kN3);
    convert_transpose_bf16<<<(nU + 255) / 256, 256, 0, stream>>>(U_fw, UtF, kU, kN3);
    convert_transpose_bf16<<<(nU + 255) / 256, 256, 0, stream>>>(U_bw, UtB, kU, kN3);
  }

  // K1: embedding gather + input projection GEMM (both directions)
  {
    dim3 grid(kM / 128, kN3 / 128, 2);
    xproj_wmma<<<grid, 256, 0, stream>>>(tokens, emb, WtF, WtB, b_in_fw,
                                         b_in_bw, xpF, xpB);
  }

  // K2: persistent bidirectional recurrence (32 WGs, LDS-resident U panels)
  {
    const size_t lds = 98304 + 32768 + 12288 + 4096;  // 147456 B
    gru_persistent<<<32, 128, lds, stream>>>(tokens, xpF, xpB, UtF, UtB,
                                             b_rec_fw, b_rec_bw, hbuf, out,
                                             bar);
  }
}
